// MambaModel_14104672600691
// MI455X (gfx1250) — matrix operand order repaired and checked by reading, compiled, not run on hardware
//
#include <hip/hip_runtime.h>
#include <hip/hip_bf16.h>
#include <math.h>

// ---------------- model constants ----------------
#define D_MODEL   1024
#define D_INNER   2048
#define D_STATE   64
#define HEADS     32
#define PDIM      64
#define KCONV     4
#define CONV_DIM  2176        // D_INNER + 2*D_STATE
#define D_IN_PROJ 4256        // 2*D_INNER + 2*D_STATE + HEADS
#define NUM_LAYERS 2
#define SEQ       2048
#define VOCAB     32000
#define CHUNK     512
#define NCHUNK    4

typedef __attribute__((ext_vector_type(16))) __bf16 v16bf;
typedef __attribute__((ext_vector_type(8)))  float  v8f;

// Pack 16 fp32 -> bf16 fragment (element order == VGPR packing order).
__device__ inline v16bf pack16(float4 f0, float4 f1, float4 f2, float4 f3) {
    v16bf r;
    r[0]=(__bf16)f0.x;  r[1]=(__bf16)f0.y;  r[2]=(__bf16)f0.z;  r[3]=(__bf16)f0.w;
    r[4]=(__bf16)f1.x;  r[5]=(__bf16)f1.y;  r[6]=(__bf16)f1.z;  r[7]=(__bf16)f1.w;
    r[8]=(__bf16)f2.x;  r[9]=(__bf16)f2.y;  r[10]=(__bf16)f2.z; r[11]=(__bf16)f2.w;
    r[12]=(__bf16)f3.x; r[13]=(__bf16)f3.y; r[14]=(__bf16)f3.z; r[15]=(__bf16)f3.w;
    return r;
}

union BPack { uint4 u; __bf16 h[8]; };
union BFrag { v16bf v; uint4 u[2]; };

// ---------------------------------------------------------------------------
// WMMA GEMM:  C[M,N] (+)= A[M,K] * B[N,K]^T  (fp32 in, bf16 WMMA, fp32 acc).
// Block = 256 threads = 8 waves.  Block tile: 128 (M) x 64 (N).
//   - wave w owns rows [tm, tm+16), all waves share the 64-wide N strip
//   - B strip (64 rows x 32 k) is staged fp32->bf16 into LDS once per K-step
//     in fragment layout; each wave reads it back with ds_load_b128
//   - each wave keeps 4 accumulators -> 4 v_wmma per K-step, A-frag reused 4x
// Requires: M % 128 == 0, K % 32 == 0, N % 16 == 0 (edge N tiles masked).
// ---------------------------------------------------------------------------
__global__ void __launch_bounds__(256)
gemm_bf16_wmma(const float* __restrict__ A, int lda,
               const float* __restrict__ B, int ldb,
               float* __restrict__ C, int ldc,
               int M, int N, int K, int addC)
{
    __shared__ __align__(32) uint4 lbs[4 * 32 * 2];   // 4 tiles x 32 lanes x 32B

    const int tid  = threadIdx.x;
    const int lane = tid & 31;
    const int wave = tid >> 5;
    const int tm = blockIdx.y * 128 + wave * 16;      // per-wave M tile
    const int tn = blockIdx.x * 64;                    // shared N strip

    const int hi = lane >> 4;
    const int mr = lane & 15;

    // --- A-fragment source: row tm+mr, K pattern {kb..kb+7, kb+16..kb+23} ---
    const float* arow = A + (size_t)(tm + mr) * lda + hi * 8;

    // --- cooperative B fill mapping: tid -> (tile j2, lane L2, half) ---
    const int j2   = tid >> 6;
    const int L2   = (tid >> 1) & 31;
    const int half = tid & 1;
    int brow_idx = tn + j2 * 16 + (L2 & 15);
    if (brow_idx > N - 1) brow_idx = N - 1;            // clamp OOB rows
    const float* gB = B + (size_t)brow_idx * ldb + (L2 >> 4) * 8 + half * 16;
    const int lfill = (j2 * 32 + L2) * 2 + half;

    v8f acc0 = {}, acc1 = {}, acc2 = {}, acc3 = {};

    for (int k0 = 0; k0 < K; k0 += 32) {
        // stage B (fp32 -> bf16, fragment layout) into LDS
        float4 f0 = *reinterpret_cast<const float4*>(gB + k0);
        float4 f1 = *reinterpret_cast<const float4*>(gB + k0 + 4);
        __builtin_prefetch(gB + k0 + 64, 0, 1);        // global_prefetch_b8
        BPack pk;
        pk.h[0] = (__bf16)f0.x; pk.h[1] = (__bf16)f0.y;
        pk.h[2] = (__bf16)f0.z; pk.h[3] = (__bf16)f0.w;
        pk.h[4] = (__bf16)f1.x; pk.h[5] = (__bf16)f1.y;
        pk.h[6] = (__bf16)f1.z; pk.h[7] = (__bf16)f1.w;
        lbs[lfill] = pk.u;

        // A fragment (private, reused by all 4 WMMAs)
        float4 a0 = *reinterpret_cast<const float4*>(arow + k0);
        float4 a1 = *reinterpret_cast<const float4*>(arow + k0 + 4);
        float4 a2 = *reinterpret_cast<const float4*>(arow + k0 + 16);
        float4 a3 = *reinterpret_cast<const float4*>(arow + k0 + 20);
        v16bf af = pack16(a0, a1, a2, a3);

        __syncthreads();
        BFrag b0, b1, b2, b3;
        b0.u[0] = lbs[(0 * 32 + lane) * 2]; b0.u[1] = lbs[(0 * 32 + lane) * 2 + 1];
        b1.u[0] = lbs[(1 * 32 + lane) * 2]; b1.u[1] = lbs[(1 * 32 + lane) * 2 + 1];
        b2.u[0] = lbs[(2 * 32 + lane) * 2]; b2.u[1] = lbs[(2 * 32 + lane) * 2 + 1];
        b3.u[0] = lbs[(3 * 32 + lane) * 2]; b3.u[1] = lbs[(3 * 32 + lane) * 2 + 1];

        acc0 = __builtin_amdgcn_wmma_f32_16x16x32_bf16(false, af, false, b0.v, (short)0, acc0, false, false);
        acc1 = __builtin_amdgcn_wmma_f32_16x16x32_bf16(false, af, false, b1.v, (short)0, acc1, false, false);
        acc2 = __builtin_amdgcn_wmma_f32_16x16x32_bf16(false, af, false, b2.v, (short)0, acc2, false, false);
        acc3 = __builtin_amdgcn_wmma_f32_16x16x32_bf16(false, af, false, b3.v, (short)0, acc3, false, false);
        __syncthreads();
    }

    // --- epilogue: branch once, mask OOB N tiles ---
    v8f accs[4] = { acc0, acc1, acc2, acc3 };
    if (addC) {
#pragma unroll
        for (int j = 0; j < 4; ++j) {
            if (tn + j * 16 >= N) continue;
#pragma unroll
            for (int r = 0; r < 8; ++r) {
                size_t off = (size_t)(tm + hi * 8 + r) * ldc + tn + j * 16 + mr;
                C[off] += accs[j][r];
            }
        }
    } else {
#pragma unroll
        for (int j = 0; j < 4; ++j) {
            if (tn + j * 16 >= N) continue;
#pragma unroll
            for (int r = 0; r < 8; ++r) {
                size_t off = (size_t)(tm + hi * 8 + r) * ldc + tn + j * 16 + mr;
                C[off] = accs[j][r];
            }
        }
    }
}

// ---------------------------------------------------------------------------
// Embedding gather: h[t,:] = embed[tokens[t],:]
// ---------------------------------------------------------------------------
__global__ void __launch_bounds__(256)
embed_kernel(const int* __restrict__ tokens, const float* __restrict__ emb,
             float* __restrict__ h)
{
    int t = blockIdx.x;
    const float* e = emb + (size_t)tokens[t] * D_MODEL;
    for (int i = threadIdx.x; i < D_MODEL; i += 256)
        h[(size_t)t * D_MODEL + i] = e[i];
}

// ---------------------------------------------------------------------------
// RMSNorm: out[t,:] = x[t,:] * rsqrt(mean(x^2)+eps) * w
// ---------------------------------------------------------------------------
__global__ void __launch_bounds__(256)
rmsnorm_kernel(const float* __restrict__ x, const float* __restrict__ w,
               float* __restrict__ out, int dim)
{
    __shared__ float red[256];
    int t = blockIdx.x;
    const float* xr = x + (size_t)t * dim;
    float s = 0.f;
    for (int i = threadIdx.x; i < dim; i += 256) { float v = xr[i]; s += v * v; }
    red[threadIdx.x] = s; __syncthreads();
    for (int o = 128; o > 0; o >>= 1) {
        if (threadIdx.x < o) red[threadIdx.x] += red[threadIdx.x + o];
        __syncthreads();
    }
    float rinv = rsqrtf(red[0] / (float)dim + 1e-5f);
    for (int i = threadIdx.x; i < dim; i += 256)
        out[(size_t)t * dim + i] = xr[i] * rinv * w[i];
}

// ---------------------------------------------------------------------------
// Depthwise causal conv (K=4) + bias + silu over xBC slice of zxbcdt.
// ---------------------------------------------------------------------------
__global__ void __launch_bounds__(256)
conv_silu_kernel(const float* __restrict__ zx, const float* __restrict__ cw,
                 const float* __restrict__ cb, float* __restrict__ xBC)
{
    int idx = blockIdx.x * 256 + threadIdx.x;
    if (idx >= SEQ * CONV_DIM) return;
    int t = idx / CONV_DIM, ch = idx % CONV_DIM;
    float acc = cb[ch];
#pragma unroll
    for (int k = 0; k < KCONV; ++k) {
        int tt = t - (KCONV - 1) + k;
        if (tt >= 0)
            acc += cw[ch * KCONV + k] * zx[(size_t)tt * D_IN_PROJ + D_INNER + ch];
    }
    xBC[idx] = acc / (1.f + expf(-acc));   // silu
}

// ---------------------------------------------------------------------------
// dt = softplus(dt_raw + dt_bias)
// ---------------------------------------------------------------------------
__global__ void __launch_bounds__(256)
dt_kernel(const float* __restrict__ zx, const float* __restrict__ dtb,
          float* __restrict__ dt)
{
    int idx = blockIdx.x * 256 + threadIdx.x;
    if (idx >= SEQ * HEADS) return;
    int t = idx / HEADS, hh = idx % HEADS;
    float x = zx[(size_t)t * D_IN_PROJ + D_INNER + CONV_DIM + hh] + dtb[hh];
    dt[idx] = (x > 20.f) ? x : log1pf(expf(x));
}

// ---------------------------------------------------------------------------
// Per (chunk, head) serial cumsum of dA = dt*A (tiny: 128 lanes x 512 steps).
// ---------------------------------------------------------------------------
__global__ void __launch_bounds__(128)
cumsum_kernel(const float* __restrict__ dt, const float* __restrict__ Alog,
              float* __restrict__ Acs, float* __restrict__ cdec)
{
    int idx = threadIdx.x;
    if (idx >= NCHUNK * HEADS) return;
    int c = idx / HEADS, hh = idx % HEADS;
    float A = -expf(Alog[hh]);
    float s = 0.f;
    for (int i = 0; i < CHUNK; ++i) {
        int t = c * CHUNK + i;
        s += dt[t * HEADS + hh] * A;
        Acs[t * HEADS + hh] = s;
    }
    cdec[idx] = expf(s);
}

// ---------------------------------------------------------------------------
// y_diag: per (chunk, head, 16-row l-tile, 16-col p-tile) wave builds the
// causal decay-weighted A-fragment in registers and WMMAs against x.
//   y[l,h,p] = sum_{s<=l} CB[l,s]*exp(Acs[l]-Acs[s])*dt[s] * x[s,h,p]
// ---------------------------------------------------------------------------
__global__ void __launch_bounds__(256)
ydiag_kernel(const float* __restrict__ CB, const float* __restrict__ Acs,
             const float* __restrict__ dt, const float* __restrict__ xBC,
             float* __restrict__ y)
{
    int lane = threadIdx.x & 31, wave = threadIdx.x >> 5;
    int task = blockIdx.x * 8 + wave;          // 4*32*32*4 = 16384 tasks
    int pt = task & 3, lt = (task >> 2) & 31, hh = (task >> 7) & 31, c = task >> 12;
    int hi = lane >> 4, mr = lane & 15;

    int lrel = lt * 16 + mr;
    int tl = c * CHUNK + lrel;
    float acsl = Acs[tl * HEADS + hh];
    int pn = pt * 16 + mr;
    const float* cbrow = CB + (size_t)c * CHUNK * CHUNK + (size_t)lrel * CHUNK;

    v8f acc = {};
    int smax = lt * 16 + 15;
    for (int k0 = 0; k0 <= smax; k0 += 32) {
        v16bf af, bf;
#pragma unroll
        for (int e = 0; e < 16; ++e) {
            int sa = k0 + hi * 8 + (e < 8 ? e : e + 8);   // A K-pattern
            float v = 0.f;
            if (sa <= lrel) {
                int ts = c * CHUNK + sa;
                v = cbrow[sa] * expf(acsl - Acs[ts * HEADS + hh]) * dt[ts * HEADS + hh];
            }
            af[e] = (__bf16)v;
            int sb = k0 + hi * 8 + (e < 8 ? e : e + 8);   // B K-pattern
            int tsb = c * CHUNK + sb;
            bf[e] = (__bf16)xBC[(size_t)tsb * CONV_DIM + hh * PDIM + pn];
        }
        acc = __builtin_amdgcn_wmma_f32_16x16x32_bf16(false, af, false, bf,
                                                      (short)0, acc, false, false);
    }
#pragma unroll
    for (int r = 0; r < 8; ++r) {
        int lr = lt * 16 + hi * 8 + r;
        y[((size_t)(c * CHUNK + lr) * HEADS + hh) * PDIM + pt * 16 + mr] = acc[r];
    }
}

// ---------------------------------------------------------------------------
// states[c,h,p,n] = sum_l x[l,h,p]*exp(Acs_last-Acs[l])*dt[l] * B[l,n]
// One wave per (c,h, p-tile, n-tile); K = l over the chunk.
// ---------------------------------------------------------------------------
__global__ void __launch_bounds__(256)
states_kernel(const float* __restrict__ Acs, const float* __restrict__ dt,
              const float* __restrict__ xBC, float* __restrict__ states)
{
    int lane = threadIdx.x & 31, wave = threadIdx.x >> 5;
    int task = blockIdx.x * 8 + wave;          // 4*32*4*4 = 2048 tasks
    int nt = task & 3, mt = (task >> 2) & 3, hh = (task >> 4) & 31, c = task >> 9;
    int hi = lane >> 4, mr = lane & 15;

    int pm = mt * 16 + mr;    // A row -> p index
    int nn = nt * 16 + mr;    // B col -> n index
    float acs_last = Acs[(c * CHUNK + CHUNK - 1) * HEADS + hh];

    v8f acc = {};
    for (int k0 = 0; k0 < CHUNK; k0 += 32) {
        v16bf af, bf;
#pragma unroll
        for (int e = 0; e < 16; ++e) {
            int la = k0 + hi * 8 + (e < 8 ? e : e + 8);
            int ta = c * CHUNK + la;
            float w = expf(acs_last - Acs[ta * HEADS + hh]) * dt[ta * HEADS + hh];
            af[e] = (__bf16)(w * xBC[(size_t)ta * CONV_DIM + hh * PDIM + pm]);
            int lb = k0 + hi * 8 + (e < 8 ? e : e + 8);
            int tb = c * CHUNK + lb;
            bf[e] = (__bf16)xBC[(size_t)tb * CONV_DIM + D_INNER + nn];
        }
        acc = __builtin_amdgcn_wmma_f32_16x16x32_bf16(false, af, false, bf,
                                                      (short)0, acc, false, false);
    }
#pragma unroll
    for (int r = 0; r < 8; ++r) {
        int pr = mt * 16 + hi * 8 + r;
        states[(((size_t)c * HEADS + hh) * PDIM + pr) * D_STATE + nt * 16 + mr] = acc[r];
    }
}

// ---------------------------------------------------------------------------
// Inter-chunk scan: prev[0]=0; prev[c] = cdec[c-1]*prev[c-1] + states[c-1]
// ---------------------------------------------------------------------------
__global__ void __launch_bounds__(256)
recur_kernel(const float* __restrict__ states, const float* __restrict__ cdec,
             float* __restrict__ prev)
{
    int idx = blockIdx.x * 256 + threadIdx.x;      // HEADS*PDIM*D_STATE
    if (idx >= HEADS * PDIM * D_STATE) return;
    int hh = idx / (PDIM * D_STATE);
    float p = 0.f;
    prev[idx] = 0.f;
    for (int c = 1; c < NCHUNK; ++c) {
        p = cdec[(c - 1) * HEADS + hh] * p
          + states[(size_t)(c - 1) * HEADS * PDIM * D_STATE + idx];
        prev[(size_t)c * HEADS * PDIM * D_STATE + idx] = p;
    }
}

// ---------------------------------------------------------------------------
// y += exp(Acs[t,h]) * (C[t,:] . prev[c,h,p,:]) + D[h]*x[t,h,p]
// ---------------------------------------------------------------------------
__global__ void __launch_bounds__(256)
yoff_kernel(const float* __restrict__ xBC, const float* __restrict__ Acs,
            const float* __restrict__ prev, const float* __restrict__ Dp,
            float* __restrict__ y)
{
    int idx = blockIdx.x * 256 + threadIdx.x;      // SEQ*HEADS*PDIM
    if (idx >= SEQ * HEADS * PDIM) return;
    int t = idx >> 11;
    int hp = idx & 2047;
    int hh = hp >> 6, p = hp & 63;
    int c = t >> 9;
    const float* Crow = xBC + (size_t)t * CONV_DIM + D_INNER + D_STATE;
    const float* pr = prev + (((size_t)c * HEADS + hh) * PDIM + p) * D_STATE;
    float acc = 0.f;
#pragma unroll 8
    for (int n = 0; n < D_STATE; ++n) acc += Crow[n] * pr[n];
    float xs = xBC[(size_t)t * CONV_DIM + hh * PDIM + p];
    y[idx] += expf(Acs[t * HEADS + hh]) * acc + Dp[hh] * xs;
}

// ---------------------------------------------------------------------------
// Gated RMSNorm: v = y * silu(z);  out = rmsnorm(v) * gw
// ---------------------------------------------------------------------------
__global__ void __launch_bounds__(256)
gatenorm_kernel(const float* __restrict__ y, const float* __restrict__ zx,
                const float* __restrict__ gw, float* __restrict__ out)
{
    __shared__ float buf[D_INNER];
    __shared__ float red[256];
    int t = blockIdx.x;
    const float* yr = y + (size_t)t * D_INNER;
    const float* zr = zx + (size_t)t * D_IN_PROJ;
    float s = 0.f;
    for (int i = threadIdx.x; i < D_INNER; i += 256) {
        float z = zr[i];
        float v = yr[i] * (z / (1.f + expf(-z)));
        buf[i] = v; s += v * v;
    }
    red[threadIdx.x] = s; __syncthreads();
    for (int o = 128; o > 0; o >>= 1) {
        if (threadIdx.x < o) red[threadIdx.x] += red[threadIdx.x + o];
        __syncthreads();
    }
    float rinv = rsqrtf(red[0] / (float)D_INNER + 1e-5f);
    for (int i = threadIdx.x; i < D_INNER; i += 256)
        out[(size_t)t * D_INNER + i] = buf[i] * rinv * gw[i];
}

// ---------------------------------------------------------------------------
extern "C" void kernel_launch(void* const* d_in, const int* in_sizes, int n_in,
                              void* d_out, int out_size, void* d_ws, size_t ws_size,
                              hipStream_t stream)
{
    (void)in_sizes; (void)n_in; (void)out_size; (void)ws_size;
    const int*   tokens  = (const int*)d_in[0];
    const float* embedw  = (const float*)d_in[1];
    const float* rms_w   = (const float*)d_in[2];
    const float* in_w    = (const float*)d_in[3];
    const float* conv_w  = (const float*)d_in[4];
    const float* conv_b  = (const float*)d_in[5];
    const float* dt_bias = (const float*)d_in[6];
    const float* A_log   = (const float*)d_in[7];
    const float* Dp      = (const float*)d_in[8];
    const float* gnorm_w = (const float*)d_in[9];
    const float* out_w   = (const float*)d_in[10];
    const float* final_w = (const float*)d_in[11];
    const float* lm_w    = (const float*)d_in[12];
    float* logits = (float*)d_out;

    // ---- workspace carving (fp32), ~112 MB total ----
    float* ws = (float*)d_ws;
    size_t o = 0;
    float* h    = ws + o; o += (size_t)SEQ * D_MODEL;
    float* xn   = ws + o; o += (size_t)SEQ * D_MODEL;
    float* zx   = ws + o; o += (size_t)SEQ * D_IN_PROJ;
    float* xBC  = ws + o; o += (size_t)SEQ * CONV_DIM;
    float* dtv  = ws + o; o += (size_t)SEQ * HEADS;
    float* Acs  = ws + o; o += (size_t)SEQ * HEADS;
    float* cdec = ws + o; o += (size_t)NCHUNK * HEADS;
    float* CB   = ws + o; o += (size_t)NCHUNK * CHUNK * CHUNK;
    float* stt  = ws + o; o += (size_t)NCHUNK * HEADS * PDIM * D_STATE;
    float* prev = ws + o; o += (size_t)NCHUNK * HEADS * PDIM * D_STATE;
    float* y    = ws + o; o += (size_t)SEQ * D_INNER;
    float* yg   = ws + o; o += (size_t)SEQ * D_INNER;

    embed_kernel<<<SEQ, 256, 0, stream>>>(tokens, embedw, h);

    for (int L = 0; L < NUM_LAYERS; ++L) {
        rmsnorm_kernel<<<SEQ, 256, 0, stream>>>(h, rms_w + (size_t)L * D_MODEL, xn, D_MODEL);

        gemm_bf16_wmma<<<dim3((D_IN_PROJ + 63) / 64, SEQ / 128), 256, 0, stream>>>(
            xn, D_MODEL, in_w + (size_t)L * D_IN_PROJ * D_MODEL, D_MODEL,
            zx, D_IN_PROJ, SEQ, D_IN_PROJ, D_MODEL, 0);

        conv_silu_kernel<<<(SEQ * CONV_DIM + 255) / 256, 256, 0, stream>>>(
            zx, conv_w + (size_t)L * CONV_DIM * KCONV, conv_b + (size_t)L * CONV_DIM, xBC);

        dt_kernel<<<(SEQ * HEADS + 255) / 256, 256, 0, stream>>>(
            zx, dt_bias + (size_t)L * HEADS, dtv);

        cumsum_kernel<<<1, 128, 0, stream>>>(dtv, A_log + (size_t)L * HEADS, Acs, cdec);

        // CB[c] = C_chunk (512x64) * B_chunk^T (64x512), shared across heads
        for (int c = 0; c < NCHUNK; ++c) {
            gemm_bf16_wmma<<<dim3(CHUNK / 64, CHUNK / 128), 256, 0, stream>>>(
                xBC + (size_t)c * CHUNK * CONV_DIM + D_INNER + D_STATE, CONV_DIM,  // C rows
                xBC + (size_t)c * CHUNK * CONV_DIM + D_INNER, CONV_DIM,            // B rows
                CB + (size_t)c * CHUNK * CHUNK, CHUNK, CHUNK, CHUNK, D_STATE, 0);
        }

        ydiag_kernel<<<(NCHUNK * HEADS * 32 * 4) / 8, 256, 0, stream>>>(CB, Acs, dtv, xBC, y);
        states_kernel<<<(NCHUNK * HEADS * 4 * 4) / 8, 256, 0, stream>>>(Acs, dtv, xBC, stt);
        recur_kernel<<<(HEADS * PDIM * D_STATE + 255) / 256, 256, 0, stream>>>(stt, cdec, prev);
        yoff_kernel<<<(SEQ * HEADS * PDIM + 255) / 256, 256, 0, stream>>>(
            xBC, Acs, prev, Dp + (size_t)L * HEADS, y);

        gatenorm_kernel<<<SEQ, 256, 0, stream>>>(y, zx, gnorm_w + (size_t)L * D_INNER, yg);

        // residual add: h += yg * out_w^T
        gemm_bf16_wmma<<<dim3(D_MODEL / 64, SEQ / 128), 256, 0, stream>>>(
            yg, D_INNER, out_w + (size_t)L * D_MODEL * D_INNER, D_INNER,
            h, D_MODEL, SEQ, D_MODEL, D_INNER, 1);
    }

    rmsnorm_kernel<<<SEQ, 256, 0, stream>>>(h, final_w, xn, D_MODEL);

    // lm_head: 2048 x 32000 logits — the 134-GFLOP finale
    gemm_bf16_wmma<<<dim3(VOCAB / 64, SEQ / 128), 256, 0, stream>>>(
        xn, D_MODEL, lm_w, D_MODEL, logits, VOCAB, SEQ, VOCAB, D_MODEL, 0);
}
